// VectorQuantizerEMA_26740466384922
// MI455X (gfx1250) — matrix operand order repaired and checked by reading, not compiled, not run on hardware
//
#include <hip/hip_runtime.h>

// ---------------- types ----------------
typedef __bf16 bf16_t;
typedef __attribute__((ext_vector_type(16))) __bf16 v16bf;
typedef __attribute__((ext_vector_type(8)))  __bf16 v8bf;
typedef __attribute__((ext_vector_type(4)))  __bf16 v4bf;
typedef __attribute__((ext_vector_type(8)))  float  v8f;
typedef __attribute__((ext_vector_type(4)))  int    v4i_t;

#define VQ_N     65536    // 16*4096 rows
#define VQ_D     512      // embedding dim
#define VQ_K     2048     // num embeddings
#define VQ_BM    64       // rows per block
#define VQ_BN    64       // codes per LDS tile
#define VQ_TILES (VQ_K / VQ_BN)   // 32
#define VQ_LS    520      // padded A row stride (bf16): 1040B = 65 quads (coprime 16)
#define VQ_BS    72       // padded B row stride (bf16): 144B  = 9 quads  (coprime 16)

static __device__ __forceinline__ v16bf cat16(v8bf a, v8bf b) {
  return __builtin_shufflevector(a, b, 0,1,2,3,4,5,6,7,8,9,10,11,12,13,14,15);
}

// ---- B-tile staging: async direct-to-LDS when available, sync copy otherwise ----
static __device__ __forceinline__ void stage_B(const bf16_t* __restrict__ WbfT,
                                               bf16_t* dst, int tile, int tid) {
#if __has_builtin(__builtin_amdgcn_global_load_async_to_lds_b128)
  #pragma unroll
  for (int i = 0; i < 16; ++i) {
    int c   = tid + i * 256;            // 4096 16B chunks: 512 depths x 64 codes
    int d   = c >> 3;
    int cin = c & 7;
    const bf16_t* g = WbfT + (size_t)d * VQ_K + tile * VQ_BN + cin * 8;
    bf16_t*       l = dst + d * VQ_BS + cin * 8;
    __builtin_amdgcn_global_load_async_to_lds_b128(
        (__attribute__((address_space(1))) v4i_t*)(v4i_t*)(void*)g,
        (__attribute__((address_space(3))) v4i_t*)(v4i_t*)l, 0, 0);
  }
#else
  #pragma unroll
  for (int i = 0; i < 16; ++i) {
    int c   = tid + i * 256;
    int d   = c >> 3;
    int cin = c & 7;
    uint4 v = *(const uint4*)(WbfT + (size_t)d * VQ_K + tile * VQ_BN + cin * 8);
    *(uint4*)((char*)dst + d * (VQ_BS * 2) + cin * 16) = v;
  }
#endif
}

static __device__ __forceinline__ void wait_async_lds() {
#if __has_builtin(__builtin_amdgcn_global_load_async_to_lds_b128)
  #if __has_builtin(__builtin_amdgcn_s_wait_asynccnt)
    __builtin_amdgcn_s_wait_asynccnt(0);
  #else
    asm volatile("s_wait_asynccnt 0x0" ::: "memory");
  #endif
#endif
}

// ---------------- kernel 1: weight -> bf16 (depth-major) + ||e||^2 ----------------
__global__ __launch_bounds__(256) void vq_prep(const float* __restrict__ W,
                                               bf16_t* __restrict__ WbfT,   // [512][2048]
                                               float* __restrict__ Wsq) {   // [2048]
  const int row = blockIdx.x;
  const int t   = threadIdx.x;
  const float* wr = W + (size_t)row * VQ_D;
  float x0 = wr[t], x1 = wr[t + 256];
  WbfT[(size_t)t * VQ_K + row]         = (bf16_t)x0;
  WbfT[(size_t)(t + 256) * VQ_K + row] = (bf16_t)x1;
  __shared__ float red[256];
  red[t] = x0 * x0 + x1 * x1;
  __syncthreads();
  for (int off = 128; off > 0; off >>= 1) {
    if (t < off) red[t] += red[t + off];
    __syncthreads();
  }
  if (t == 0) Wsq[row] = red[0];
}

// ---------------- kernel 2: WMMA distance + argmin (double-buffered B) ----------------
__global__ __launch_bounds__(256) void vq_argmin(const float* __restrict__ X,
                                                 const bf16_t* __restrict__ WbfT,
                                                 const float* __restrict__ Wsq,
                                                 int* __restrict__ outIdx) {
  __shared__ bf16_t Abuf[VQ_BM * VQ_LS];      // 66,560 B : [row][depth]
  __shared__ bf16_t Bbuf[2][VQ_D * VQ_BS];    // 2 x 73,728 B : [depth][code]
  __shared__ float  redV[8 * 16];
  __shared__ int    redI[8 * 16];

  const int tid     = threadIdx.x;
  const int lane    = tid & 31;
  const int wave    = tid >> 5;
  const int l15     = lane & 15;
  const int lHi     = lane >> 4;
  const int wRow    = (wave >> 1) * 16;
  const int wCol    = (wave & 1) * 32;
  const int rowBase = blockIdx.x * VQ_BM;

  // ---- stage A: 64 rows x 512 f32 -> bf16 LDS (needs conversion; stays sync)
  {
    const float4* Xv = (const float4*)(X + (size_t)rowBase * VQ_D);
    #pragma unroll
    for (int i = 0; i < 32; ++i) {
      int c   = tid + i * 256;
      int r   = c >> 7;
      int cin = c & 127;
      float4 v = Xv[c];
      v4bf b; b.x = (bf16_t)v.x; b.y = (bf16_t)v.y; b.z = (bf16_t)v.z; b.w = (bf16_t)v.w;
      *(v4bf*)&Abuf[r * VQ_LS + cin * 4] = b;
    }
  }

  // prefetch first B tile while A stores drain
  stage_B(WbfT, &Bbuf[0][0], 0, tid);

  float bestV[8];
  int   bestI[8];
  #pragma unroll
  for (int r = 0; r < 8; ++r) { bestV[r] = 3.4e38f; bestI[r] = 0; }

  for (int ct = 0; ct < VQ_TILES; ++ct) {
    const int cur = ct & 1;
    wait_async_lds();        // this wave's slice of tile ct has landed in LDS
    __syncthreads();         // whole tile ct visible; buffer cur^1 reads (tile ct-1) done

    if (ct + 1 < VQ_TILES)   // overlap next tile's DMA with this tile's WMMAs
      stage_B(WbfT, &Bbuf[cur ^ 1][0], ct + 1, tid);

    const bf16_t* Bcur = &Bbuf[cur][0];
    v8f acc0 = {};
    v8f acc1 = {};
    #pragma unroll 4
    for (int dk = 0; dk < VQ_D; dk += 32) {
      // A fragment: lane = row M=l15; elems 0-7: K=dk+8*lHi.., elems 8-15: K=dk+16+8*lHi..
      const bf16_t* arow = &Abuf[(wRow + l15) * VQ_LS + dk + 8 * lHi];
      v16bf av = cat16(*(const v8bf*)arow, *(const v8bf*)(arow + 16));
      // B fragments: lane = code N=l15; elems 0-7: K=dk+8*lHi.., elems 8-15: K=dk+16+8*lHi..
      const bf16_t* bcol = Bcur + (dk + 8 * lHi) * VQ_BS + wCol + l15;
      v16bf bv0, bv1;
      #pragma unroll
      for (int s = 0; s < 16; ++s) {
        int kk = (s < 8) ? s : s + 8;
        bv0[s] = bcol[kk * VQ_BS];
        bv1[s] = bcol[kk * VQ_BS + 16];
      }
      acc0 = __builtin_amdgcn_wmma_f32_16x16x32_bf16(false, av, false, bv0,
                                                     (short)0, acc0, false, false);
      acc1 = __builtin_amdgcn_wmma_f32_16x16x32_bf16(false, av, false, bv1,
                                                     (short)0, acc1, false, false);
    }

    // distance = ||e||^2 - 2 x.e ; running argmin (cols ascend -> strict '<' keeps first)
    {
      int   c0 = ct * VQ_BN + wCol + l15;
      int   c1 = c0 + 16;
      float q0 = Wsq[c0];
      float q1 = Wsq[c1];
      #pragma unroll
      for (int r = 0; r < 8; ++r) {
        float d0 = q0 - 2.0f * acc0[r];
        if (d0 < bestV[r]) { bestV[r] = d0; bestI[r] = c0; }
        float d1 = q1 - 2.0f * acc1[r];
        if (d1 < bestV[r]) { bestV[r] = d1; bestI[r] = c1; }
      }
    }
  }

  // ---- reduce across the 16 lanes of each half (C/D: elem r -> M=r+8*lHi, N=l15)
  #pragma unroll
  for (int r = 0; r < 8; ++r) {
    float v  = bestV[r];
    int   ix = bestI[r];
    #pragma unroll
    for (int off = 8; off > 0; off >>= 1) {
      float v2 = __shfl_xor(v, off, 32);
      int   i2 = __shfl_xor(ix, off, 32);
      if (v2 < v || (v2 == v && i2 < ix)) { v = v2; ix = i2; }
    }
    bestV[r] = v; bestI[r] = ix;
  }
  if (l15 == 0) {
    #pragma unroll
    for (int r = 0; r < 8; ++r) {
      int rr = r + 8 * lHi;
      redV[wave * 16 + rr] = bestV[r];
      redI[wave * 16 + rr] = bestI[r];
    }
  }
  __syncthreads();
  if (tid < VQ_BM) {
    int row = tid, g = row >> 4, rr = row & 15;
    float vA = redV[(2 * g) * 16 + rr];     int iA = redI[(2 * g) * 16 + rr];
    float vB = redV[(2 * g + 1) * 16 + rr]; int iB = redI[(2 * g + 1) * 16 + rr];
    bool sel = (vB < vA) || (vB == vA && iB < iA);
    outIdx[rowBase + row] = sel ? iB : iA;
  }
}

// ---------------- kernel 3: gather quantized, one-hot encodings, loss partials ----------------
__global__ __launch_bounds__(256) void vq_gather(const float* __restrict__ X,
                                                 const float* __restrict__ W,
                                                 const int* __restrict__ idx,
                                                 float* __restrict__ outQ,
                                                 float* __restrict__ outE,
                                                 float* __restrict__ partial) {
  const int n = blockIdx.x;
  const int t = threadIdx.x;
  const int k = idx[n];
  const float* wr = W + (size_t)k * VQ_D;
  const float* xr = X + (size_t)n * VQ_D;
  float* qr = outQ + (size_t)n * VQ_D;
  float s = 0.f;
  #pragma unroll
  for (int i = 0; i < 2; ++i) {
    int c = t + i * 256;
    float w = wr[c], x = xr[c];
    qr[c] = w;
    float d = w - x;
    s += d * d;
  }
  float* er = outE + (size_t)n * VQ_K;        // base only 4B-aligned; scalar stores
  #pragma unroll
  for (int i = 0; i < 8; ++i) {
    int c = t + i * 256;
    er[c] = (c == k) ? 1.0f : 0.0f;
  }
  __shared__ float red[256];
  red[t] = s;
  __syncthreads();
  for (int off = 128; off > 0; off >>= 1) {
    if (t < off) red[t] += red[t + off];
    __syncthreads();
  }
  if (t == 0) partial[n] = red[0];
}

// ---------------- kernel 4: finalize loss (deterministic, fixed order) ----------------
__global__ __launch_bounds__(256) void vq_loss(const float* __restrict__ partial,
                                               float* __restrict__ lossOut) {
  __shared__ float red[256];
  float s = 0.f;
  for (int i = threadIdx.x; i < VQ_N; i += 256) s += partial[i];
  red[threadIdx.x] = s;
  __syncthreads();
  for (int off = 128; off > 0; off >>= 1) {
    if (threadIdx.x < off) red[threadIdx.x] += red[threadIdx.x + off];
    __syncthreads();
  }
  if (threadIdx.x == 0)
    *lossOut = 0.25f * red[0] / (float)((size_t)VQ_N * VQ_D);
}

// ---------------- launch ----------------
extern "C" void kernel_launch(void* const* d_in, const int* in_sizes, int n_in,
                              void* d_out, int out_size, void* d_ws, size_t ws_size,
                              hipStream_t stream) {
  const float* X = (const float*)d_in[0];    // [16,4096,512]
  const float* W = (const float*)d_in[1];    // [2048,512]

  float* outQ = (float*)d_out;                               // 33,554,432 floats
  float* loss = (float*)d_out + (size_t)VQ_N * VQ_D;         // 1 float
  float* outE = loss + 1;                                    // 134,217,728 floats

  char* ws = (char*)d_ws;
  bf16_t* WbfT = (bf16_t*)ws;                                          // 2 MB
  float*  Wsq  = (float*)(ws + (size_t)2 * 1024 * 1024);               // 8 KB
  int*    Idx  = (int*)(ws + (size_t)2 * 1024 * 1024 + 8192);          // 256 KB
  float*  Part = (float*)(ws + (size_t)2 * 1024 * 1024 + 8192 + 262144); // 256 KB

  vq_prep  <<<VQ_K,         256, 0, stream>>>(W, WbfT, Wsq);
  vq_argmin<<<VQ_N / VQ_BM, 256, 0, stream>>>(X, WbfT, Wsq, Idx);
  vq_gather<<<VQ_N,         256, 0, stream>>>(X, W, Idx, outQ, outE, Part);
  vq_loss  <<<1,            256, 0, stream>>>(Part, loss);
}
